// NestedSwiGLUMLP_68298569941561
// MI455X (gfx1250) — matrix operand order repaired and checked by reading, not compiled, not run on hardware
//
#include <hip/hip_runtime.h>
#include <hip/hip_bf16.h>
#include <stdint.h>

#define T_TOK 32768
#define DDIM  1024
#define HDIM  2048
#define HH    1024
#define MTILE 64
#define LDS_STRIDE 1032   // bf16 elements, padded for bank-conflict-free b128 reads

typedef __attribute__((ext_vector_type(16))) __bf16        v16bf;
typedef __attribute__((ext_vector_type(8)))  float         v8f;
typedef __attribute__((ext_vector_type(4)))  float         v4f;
typedef __attribute__((ext_vector_type(8)))  unsigned short u16x8;
typedef __attribute__((ext_vector_type(4)))  unsigned short u16x4;

union FragU { v16bf bf; u16x8 h[2]; };

__device__ __forceinline__ unsigned short f2bf(float f) {
  unsigned u = __builtin_bit_cast(unsigned, f);
  u += 0x7FFFu + ((u >> 16) & 1u);           // round-to-nearest-even
  return (unsigned short)(u >> 16);
}

__device__ __forceinline__ v8f wmma_bf16(v16bf a, v16bf b, v8f c) {
  return __builtin_amdgcn_wmma_f32_16x16x32_bf16(
      /*neg_a=*/false, a, /*neg_b=*/false, b,
      /*c_mod=*/(short)0, c, /*reuse_a=*/false, /*reuse_b=*/false);
}

// A-frag 16x32 bf16: rowPtr already includes (rb + lane%16)*ld ; s = lane>>4
// lanes 0-15: K {k0..k0+7},{k0+16..k0+23}; lanes 16-31: +8
__device__ __forceinline__ v16bf loadA(const unsigned short* rowPtr, int k0, int s) {
  FragU f;
  f.h[0] = *(const u16x8*)(rowPtr + k0 + s * 8);
  f.h[1] = *(const u16x8*)(rowPtr + k0 + 16 + s * 8);
  return f.bf;
}
// B-frag 32x16: colPtr already includes (nb + lane%16)*ld ; same K layout as A-frag
__device__ __forceinline__ v16bf loadB(const unsigned short* colPtr, int k0, int s) {
  FragU f;
  const unsigned short* p = colPtr + k0 + s * 8;
  f.h[0] = *(const u16x8*)(p);
  f.h[1] = *(const u16x8*)(p + 16);
  return f.bf;
}

// ---------------- binning (count-sort by expert) ----------------
__global__ void k_init(int* cnt, int* off, int* fill) {
  int i = threadIdx.x;
  if (i < 4) { cnt[i] = 0; off[i] = 0; fill[i] = 0; }
}
__global__ void k_count(const int* __restrict__ mask, int* cnt) {
  int t = blockIdx.x * 256 + threadIdx.x;
  atomicAdd(&cnt[mask[t]], 1);
}
__global__ void k_scan(const int* __restrict__ cnt, int* off) {
  if (threadIdx.x == 0) {
    int a = 0;
    for (int e = 0; e < 4; ++e) { off[e] = a; a += cnt[e]; }
  }
}
__global__ void k_place(const int* __restrict__ mask, const int* __restrict__ off,
                        int* fill, int* __restrict__ perm) {
  int t = blockIdx.x * 256 + threadIdx.x;
  int e = mask[t];
  int pos = off[e] + atomicAdd(&fill[e], 1);
  perm[pos] = t;
}

// ---------------- weight conversion to bf16 ----------------
// w1b[h][d] = bf16(w1[h][d]) (2048x1024); w2b[d][h] = bf16(w2[d][h<1024]) (1024x1024)
__global__ void k_cvtw(const float* __restrict__ w1, const float* __restrict__ w2,
                       unsigned short* __restrict__ w1b, unsigned short* __restrict__ w2b) {
  long idx = (long)blockIdx.x * 256 + threadIdx.x;
  const long n1 = (long)HDIM * DDIM;
  if (idx < n1) {
    w1b[idx] = f2bf(w1[idx]);
  } else {
    long i = idx - n1;                      // < 1024*1024
    int d = (int)(i >> 10), h = (int)(i & 1023);
    w2b[i] = f2bf(w2[(long)d * HDIM + h]);
  }
}

// ---------------- GEMM1: z = x*w1^T, SwiGLU, nested mask, write a (bf16) ----------------
__global__ void __launch_bounds__(256) k_gemm1(
    const float* __restrict__ x, const int* __restrict__ mask,
    const unsigned short* __restrict__ w1b, const float* __restrict__ b1,
    const int* __restrict__ perm, unsigned short* __restrict__ aB) {
  extern __shared__ unsigned short lds[];   // MTILE * LDS_STRIDE bf16
  __shared__ int s_tok[MTILE];
  __shared__ int s_act[MTILE];

  const int tid = threadIdx.x;
  const int lane = tid & 31, wave = tid >> 5;
  const int lane16 = lane & 15, s = lane >> 4;
  const int mw = wave & 1, nw = wave >> 1;
  const int blk = blockIdx.x;

  if (tid < MTILE) {
    int t = perm[blk * MTILE + tid];
    s_tok[tid] = t;
    s_act[tid] = 128 << mask[t];
  }
  __syncthreads();
  const int tileActive = s_act[MTILE - 1];  // experts sorted ascending -> max

  // stage x tile -> LDS as bf16
  for (int r = wave; r < MTILE; r += 8) {
    const float* src = x + (size_t)s_tok[r] * DDIM;
    unsigned short* dst = lds + r * LDS_STRIDE;
    for (int c = lane * 4; c < DDIM; c += 128) {
      v4f v = *(const v4f*)(src + c);
      u16x4 o = { f2bf(v.x), f2bf(v.y), f2bf(v.z), f2bf(v.w) };
      *(u16x4*)(dst + c) = o;
    }
  }
  __syncthreads();

  const unsigned short* aRow0 = lds + (mw * 32 + lane16) * LDS_STRIDE;
  const unsigned short* aRow1 = aRow0 + 16 * LDS_STRIDE;

  const int nIter = tileActive >> 7;        // 128 cols of 'a' per iter (4 nw * 32)
  for (int it = 0; it < nIter; ++it) {
    const int nbase = it * 128 + nw * 32;
    v8f Cg[2][2] = {};
    v8f Cu[2][2] = {};
    const unsigned short* bg0 = w1b + (size_t)(nbase + lane16) * DDIM;
    const unsigned short* bg1 = bg0 + (size_t)16 * DDIM;
    const unsigned short* bu0 = bg0 + (size_t)HH * DDIM;
    const unsigned short* bu1 = bg1 + (size_t)HH * DDIM;

    for (int k = 0; k < DDIM; k += 32) {
      v16bf A0  = loadA(aRow0, k, s);
      v16bf A1  = loadA(aRow1, k, s);
      v16bf Bg0 = loadB(bg0, k, s);
      v16bf Bg1 = loadB(bg1, k, s);
      v16bf Bu0 = loadB(bu0, k, s);
      v16bf Bu1 = loadB(bu1, k, s);
      Cg[0][0] = wmma_bf16(A0, Bg0, Cg[0][0]);
      Cg[0][1] = wmma_bf16(A0, Bg1, Cg[0][1]);
      Cg[1][0] = wmma_bf16(A1, Bg0, Cg[1][0]);
      Cg[1][1] = wmma_bf16(A1, Bg1, Cg[1][1]);
      Cu[0][0] = wmma_bf16(A0, Bu0, Cu[0][0]);
      Cu[0][1] = wmma_bf16(A0, Bu1, Cu[0][1]);
      Cu[1][0] = wmma_bf16(A1, Bu0, Cu[1][0]);
      Cu[1][1] = wmma_bf16(A1, Bu1, Cu[1][1]);
    }

    #pragma unroll
    for (int tm = 0; tm < 2; ++tm) {
      #pragma unroll
      for (int tn = 0; tn < 2; ++tn) {
        const int c = nbase + tn * 16 + lane16;
        const float bg = b1[c];
        const float bu = b1[HH + c];
        #pragma unroll
        for (int r = 0; r < 8; ++r) {
          const int row = mw * 32 + tm * 16 + r + 8 * s;  // C layout: VGPR r -> M=r(+8)
          float g = Cg[tm][tn][r] + bg;
          float u = Cu[tm][tn][r] + bu;
          float sig = 1.0f / (1.0f + __expf(-g));
          float val = (c < s_act[row]) ? g * sig * u : 0.0f;
          aB[(size_t)(blk * MTILE + row) * HH + c] = f2bf(val);
        }
      }
    }
  }
}

// ---------------- GEMM2: out = a * w2b^T + b2, scatter rows by token ----------------
__global__ void __launch_bounds__(256) k_gemm2(
    const unsigned short* __restrict__ aB, const unsigned short* __restrict__ w2b,
    const float* __restrict__ b2, const int* __restrict__ mask,
    const int* __restrict__ perm, float* __restrict__ out) {
  extern __shared__ unsigned short lds[];
  __shared__ int s_tok[MTILE];
  __shared__ int s_act[MTILE];

  const int tid = threadIdx.x;
  const int lane = tid & 31, wave = tid >> 5;
  const int lane16 = lane & 15, s = lane >> 4;
  const int mw = wave & 1, nw = wave >> 1;
  const int blk = blockIdx.x;

  if (tid < MTILE) {
    int t = perm[blk * MTILE + tid];
    s_tok[tid] = t;
    s_act[tid] = 128 << mask[t];
  }
  __syncthreads();
  const int tileActive = s_act[MTILE - 1];

  // stage a tile (only active K range) -> LDS
  for (int r = wave; r < MTILE; r += 8) {
    const unsigned short* src = aB + (size_t)(blk * MTILE + r) * HH;
    unsigned short* dst = lds + r * LDS_STRIDE;
    for (int c = lane * 8; c < tileActive; c += 256)
      *(u16x8*)(dst + c) = *(const u16x8*)(src + c);
  }
  __syncthreads();

  const unsigned short* aRow0 = lds + (mw * 32 + lane16) * LDS_STRIDE;
  const unsigned short* aRow1 = aRow0 + 16 * LDS_STRIDE;

  for (int it = 0; it < 8; ++it) {          // full N = 1024 output cols
    const int nbase = it * 128 + nw * 32;
    v8f C[2][2] = {};
    const unsigned short* wb0 = w2b + (size_t)(nbase + lane16) * HH;
    const unsigned short* wb1 = wb0 + (size_t)16 * HH;

    for (int k = 0; k < tileActive; k += 32) {
      v16bf A0 = loadA(aRow0, k, s);
      v16bf A1 = loadA(aRow1, k, s);
      v16bf B0 = loadB(wb0, k, s);
      v16bf B1 = loadB(wb1, k, s);
      C[0][0] = wmma_bf16(A0, B0, C[0][0]);
      C[0][1] = wmma_bf16(A0, B1, C[0][1]);
      C[1][0] = wmma_bf16(A1, B0, C[1][0]);
      C[1][1] = wmma_bf16(A1, B1, C[1][1]);
    }

    #pragma unroll
    for (int tm = 0; tm < 2; ++tm) {
      #pragma unroll
      for (int tn = 0; tn < 2; ++tn) {
        const int c = nbase + tn * 16 + lane16;
        const float bb = b2[c];
        #pragma unroll
        for (int r = 0; r < 8; ++r) {
          const int row = mw * 32 + tm * 16 + r + 8 * s;
          const int tok = s_tok[row];
          out[(size_t)tok * DDIM + c] = C[tm][tn][r] + bb;
        }
      }
    }
  }
}

extern "C" void kernel_launch(void* const* d_in, const int* in_sizes, int n_in,
                              void* d_out, int out_size, void* d_ws, size_t ws_size,
                              hipStream_t stream) {
  const float* x  = (const float*)d_in[0];
  const int* mask = (const int*)d_in[1];
  const float* w1 = (const float*)d_in[2];
  const float* b1 = (const float*)d_in[3];
  const float* w2 = (const float*)d_in[4];
  const float* b2 = (const float*)d_in[5];
  float* out = (float*)d_out;

  char* ws = (char*)d_ws;
  unsigned short* w1b = (unsigned short*)ws;                 // 4 MB  (2048x1024 bf16)
  unsigned short* w2b = (unsigned short*)(ws + 4194304);     // 2 MB  (1024x1024 bf16)
  unsigned short* aB  = (unsigned short*)(ws + 6291456);     // 64 MB (32768x1024 bf16)
  int* perm = (int*)(ws + 73400320);                         // 128 KB
  int* cnt  = (int*)(ws + 73531392);
  int* off  = cnt + 4;
  int* fill = cnt + 8;

  const int smem = MTILE * LDS_STRIDE * 2;                   // 132096 B (< 320 KB/WGP)
  (void)hipFuncSetAttribute((const void*)k_gemm1,
                            hipFuncAttributeMaxDynamicSharedMemorySize, smem);
  (void)hipFuncSetAttribute((const void*)k_gemm2,
                            hipFuncAttributeMaxDynamicSharedMemorySize, smem);

  k_init <<<1, 64, 0, stream>>>(cnt, off, fill);
  k_count<<<T_TOK / 256, 256, 0, stream>>>(mask, cnt);
  k_scan <<<1, 32, 0, stream>>>(cnt, off);
  k_place<<<T_TOK / 256, 256, 0, stream>>>(mask, off, fill, perm);
  k_cvtw <<<(HDIM * DDIM + HH * DDIM) / 256, 256, 0, stream>>>(w1, w2, w1b, w2b);
  k_gemm1<<<T_TOK / MTILE, 256, smem, stream>>>(x, mask, w1b, b1, perm, aB);
  k_gemm2<<<T_TOK / MTILE, 256, smem, stream>>>(aB, w2b, b2, mask, perm, out);
}
